// Network_1400159338853
// MI455X (gfx1250) — matrix operand order repaired and checked by reading, not compiled, not run on hardware
//
#include <hip/hip_runtime.h>

typedef __attribute__((ext_vector_type(16))) _Float16 v16h;
typedef __attribute__((ext_vector_type(8)))  _Float16 v8h;
typedef __attribute__((ext_vector_type(8)))  float    v8f;

#define THETA 10.0f
#define TS 300
#define TP 304               // T padded to 16 for WMMA N-tiles
#define KSRM 77              // truncated alpha kernel length (tau=10, eps=0.01)
#define KREF 11              // refractory kernel length (tau=1, mult=-20)

__device__ __forceinline__ float srm_tap(int s) {
    float t = (float)s;
    return t * 0.1f * __expf(1.0f - t * 0.1f);
}
__device__ __forceinline__ float ref_tap(int s) {
    float t = (float)s;
    return -2.0f * THETA * t * __expf(1.0f - t);
}

// ---------------------------------------------------------------------------
// Causal FIR along T: y[r,t] = sum_s x[r,t-s]*SRM[s].  One block per row.
// ---------------------------------------------------------------------------
__global__ void psp_fir(const float* __restrict__ x, float* __restrict__ y) {
    __shared__ float xs[TS];
    __shared__ float srm[KSRM];
    size_t row = blockIdx.x;
    const float* xr = x + row * TS;
    for (int t = threadIdx.x; t < TS; t += blockDim.x) xs[t] = xr[t];
    for (int s = threadIdx.x; s < KSRM; s += blockDim.x) srm[s] = srm_tap(s);
    __syncthreads();
    float* yr = y + row * TS;
    for (int t = threadIdx.x; t < TS; t += blockDim.x) {
        float acc = 0.0f;
        int smax = t < KSRM - 1 ? t : KSRM - 1;
        for (int s = 1; s <= smax; ++s) acc += xs[t - s] * srm[s];  // SRM[0]==0
        yr[t] = acc;
    }
}

// ---------------------------------------------------------------------------
// pool2 then FIR (they commute: both linear).  Input (NC,H,W,T) spikes,
// output rows over (NC,H/2,W/2).  Pool weight = 1.1*theta/4 = 2.75.
// ---------------------------------------------------------------------------
__global__ void pool_psp(const float* __restrict__ x, float* __restrict__ y,
                         int H, int W) {
    __shared__ float xs[TS];
    __shared__ float srm[KSRM];
    int H2 = H >> 1, W2 = W >> 1;
    size_t row = blockIdx.x;
    int w2 = (int)(row % W2);
    int h2 = (int)((row / W2) % H2);
    size_t nc = row / ((size_t)W2 * H2);
    const float* b0 = x + (((nc * H + 2 * h2) * W) + 2 * w2) * TS;
    const float* b1 = b0 + TS;
    const float* b2 = b0 + (size_t)W * TS;
    const float* b3 = b2 + TS;
    for (int t = threadIdx.x; t < TS; t += blockDim.x)
        xs[t] = 2.75f * (b0[t] + b1[t] + b2[t] + b3[t]);
    for (int s = threadIdx.x; s < KSRM; s += blockDim.x) srm[s] = srm_tap(s);
    __syncthreads();
    float* yr = y + row * TS;
    for (int t = threadIdx.x; t < TS; t += blockDim.x) {
        float acc = 0.0f;
        int smax = t < KSRM - 1 ? t : KSRM - 1;
        for (int s = 1; s <= smax; ++s) acc += xs[t - s] * srm[s];
        yr[t] = acc;
    }
}

// ---------------------------------------------------------------------------
// conv1 (16,2,5,5 pad=1 on 34x34 -> 32x32) fused with the spike scan.
// One thread per output neuron; refractory ring in registers.
// ---------------------------------------------------------------------------
__global__ void conv1_spike(const float* __restrict__ p0,
                            const float* __restrict__ w1,
                            float* __restrict__ s1) {
    int idx = blockIdx.x * blockDim.x + threadIdx.x;      // (n,co,h,w)
    if (idx >= 8 * 16 * 32 * 32) return;
    int w  = idx & 31;
    int h  = (idx >> 5) & 31;
    int co = (idx >> 10) & 15;
    int n  = idx >> 14;
    float wr[50];
    #pragma unroll
    for (int i = 0; i < 50; ++i) wr[i] = w1[co * 50 + i];
    float refc[KREF], ring[KREF];
    #pragma unroll
    for (int k = 0; k < KREF; ++k) { refc[k] = ref_tap(k); ring[k] = 0.0f; }
    float* srow = s1 + (size_t)idx * TS;
    for (int t = 0; t < TS; ++t) {
        float u = ring[0];
        #pragma unroll
        for (int ci = 0; ci < 2; ++ci)
            #pragma unroll
            for (int kh = 0; kh < 5; ++kh) {
                int yy = h + kh - 1;
                if (yy < 0 || yy >= 34) continue;
                const float* prow = p0 + (((size_t)(n * 2 + ci) * 34 + yy) * 34) * TS;
                #pragma unroll
                for (int kw = 0; kw < 5; ++kw) {
                    int xx = w + kw - 1;
                    if (xx < 0 || xx >= 34) continue;
                    u += prow[(size_t)xx * TS + t] * wr[(ci * 5 + kh) * 5 + kw];
                }
            }
        float sp = (u >= THETA) ? 1.0f : 0.0f;
        #pragma unroll
        for (int k = 0; k < KREF - 1; ++k) ring[k] = ring[k + 1] + sp * refc[k + 1];
        ring[KREF - 1] = 0.0f;
        srow[t] = sp;
    }
}

// ---------------------------------------------------------------------------
// Spike scan with refractory feedback; safe in place (u == s allowed).
// ---------------------------------------------------------------------------
__global__ void spike_scan(const float* __restrict__ u, float* __restrict__ s,
                           int rows) {
    int r = blockIdx.x * blockDim.x + threadIdx.x;
    if (r >= rows) return;
    const float* ur = u + (size_t)r * TS;
    float* sr = s + (size_t)r * TS;
    float refc[KREF], ring[KREF];
    #pragma unroll
    for (int k = 0; k < KREF; ++k) { refc[k] = ref_tap(k); ring[k] = 0.0f; }
    for (int t = 0; t < TS; ++t) {
        float ue = ur[t] + ring[0];
        float sp = (ue >= THETA) ? 1.0f : 0.0f;
        #pragma unroll
        for (int k = 0; k < KREF - 1; ++k) ring[k] = ring[k + 1] + sp * refc[k + 1];
        ring[KREF - 1] = 0.0f;
        sr[t] = sp;
    }
}

// ---------------------------------------------------------------------------
// Pack activations f32 (n,c,h,w,t) -> f16 (n,h,w,TP,c), zero pad t in [300,304).
// Makes each WMMA B-lane's 16 K-halves one contiguous 32B load.
// ---------------------------------------------------------------------------
__global__ void pack_act(const float* __restrict__ p, _Float16* __restrict__ ph,
                         int C, int H, int W) {
    size_t row = blockIdx.x;                      // over (n,c,h,w)
    int w = (int)(row % W);
    int h = (int)((row / W) % H);
    int c = (int)((row / ((size_t)W * H)) % C);
    int n = (int)(row / ((size_t)W * H * C));
    const float* src = p + row * TS;
    _Float16* dst = ph + (((size_t)(n * H + h) * W + w) * TP) * C + c;
    for (int t = threadIdx.x; t < TP; t += blockDim.x)
        dst[(size_t)t * C] = (t < TS) ? (_Float16)src[t] : (_Float16)0.0f;
}

// ---------------------------------------------------------------------------
// Pack weights f32 [Co][Ci][KH][KW] -> f16 [CoP][KP], K = tap*Ci + ci,
// zero padded so the GEMM needs no K guards on the A side.
// ---------------------------------------------------------------------------
__global__ void pack_w(const float* __restrict__ w, _Float16* __restrict__ wp,
                       int Co, int Ci, int KH, int KW, int CoP, int KP) {
    int i = blockIdx.x * blockDim.x + threadIdx.x;
    if (i >= CoP * KP) return;
    int k = i % KP;
    int co = i / KP;
    int K = Ci * KH * KW;
    float v = 0.0f;
    if (co < Co && k < K) {
        int ci = k % Ci;
        int tap = k / Ci;
        int kh = tap / KW, kw = tap % KW;
        v = w[(((size_t)co * Ci + ci) * KH + kh) * KW + kw];
    }
    wp[i] = (_Float16)v;
}

// ---------------------------------------------------------------------------
// Implicit-GEMM conv via v_wmma_f32_16x16x32_f16, templated on geometry so
// all tap/ci decompositions fold to constants and the K-chunk loop fully
// unrolls into a static stream of back-to-back WMMAs.
// Tile: M=16 output channels x N=16 timesteps, K chunked by 32 over (tap,ci).
// One wave per tile; tile guard is wave-uniform so EXEC is all ones at WMMA.
// ---------------------------------------------------------------------------
template <int Ci, int H, int W, int Co, int CoP, int KH, int KW, int PAD,
          int OH, int OW, int KP>
__launch_bounds__(256)
__global__ void conv_wmma(const _Float16* __restrict__ ph,  // [N][H][W][TP][Ci]
                          const _Float16* __restrict__ wp,  // [CoP][KP]
                          float* __restrict__ uout,         // [N][Co][OH][OW][TS]
                          int nTiles) {
    const int wg = blockIdx.x * (blockDim.x >> 5) + (threadIdx.x >> 5);
    if (wg >= nTiles) return;                     // wave-uniform
    const int lane = threadIdx.x & 31;
    const int hi = lane >> 4;                     // 0: lanes 0-15, 1: 16-31
    const int lo = lane & 15;

    constexpr int TT = 19;                        // ceil(300/16)
    constexpr int K = Ci * KH * KW;
    constexpr int NCHUNK = KP / 32;
    int tt   = wg % TT;
    int cob  = (wg / TT) % (CoP / 16);
    int site = wg / (TT * (CoP / 16));
    int x = site % OW;
    int y = (site / OW) % OH;
    int n = site / (OW * OH);
    const int t = tt * 16 + lo;

    v8f acc = {0.f, 0.f, 0.f, 0.f, 0.f, 0.f, 0.f, 0.f};
    const _Float16* wrow = wp + (size_t)(cob * 16 + lo) * KP;

    #pragma unroll
    for (int c = 0; c < NCHUNK; ++c) {
        // A fragment (weights): halves 0..7 <-> K=base..base+7,
        //                       halves 8..15 <-> K=base+8..base+15
        int ka = (c << 5) + (hi << 4);
        v8h a0 = *(const v8h*)(wrow + ka);
        v8h a1 = *(const v8h*)(wrow + ka + 8);
        v16h a;
        #pragma unroll
        for (int e = 0; e < 8; ++e) { a[e] = a0[e]; a[8 + e] = a1[e]; }
        if (c + 1 < NCHUNK)                       // CDNA5 global_prefetch path
            __builtin_prefetch(wrow + ((c + 1) << 5), 0, 3);

        // B fragment (activations): lane holds K = g0..g0+15 for its column t.
        // Because Ci is a multiple of 16, each run stays inside one tap ->
        // 16 contiguous f16 (32B) in the packed layout.
        v16h b;
        int g0 = (c << 5) + (hi << 4);
        bool ok = (g0 < K);
        int tap = g0 / Ci;                        // compile-time divisor
        int ci  = g0 % Ci;
        int kh = tap / KW, kw = tap % KW;
        int yy = y + kh - PAD, xx = x + kw - PAD;
        ok = ok && (yy >= 0) && (yy < H) && (xx >= 0) && (xx < W);
        if (ok) {
            const _Float16* src =
                ph + ((((size_t)(n * H + yy) * W + xx) * TP + t) * Ci + ci);
            b = *(const v16h*)src;
        } else {
            #pragma unroll
            for (int e = 0; e < 16; ++e) b[e] = (_Float16)0.0f;
        }

        acc = __builtin_amdgcn_wmma_f32_16x16x32_f16(
            false, a, false, b, (short)0, acc, false, false);
    }

    if (t < TS) {
        #pragma unroll
        for (int j = 0; j < 8; ++j) {
            int co = cob * 16 + j + (hi << 3);
            if (co < Co)
                uout[((((size_t)n * Co + co) * OH + y) * OW + x) * TS + t] = acc[j];
        }
    }
}

// ---------------------------------------------------------------------------
// Orchestration.  Arena A (40MB) and B (160MB) reused across stage lifetimes.
// ---------------------------------------------------------------------------
extern "C" void kernel_launch(void* const* d_in, const int* in_sizes, int n_in,
                              void* d_out, int out_size, void* d_ws, size_t ws_size,
                              hipStream_t stream) {
    const float* x   = (const float*)d_in[0];   // (8,2,34,34,300)
    const float* w1  = (const float*)d_in[1];   // (16,2,5,5,1)
    const float* w2  = (const float*)d_in[2];   // (32,16,3,3,1)
    const float* w3  = (const float*)d_in[3];   // (64,32,3,3,1)
    const float* wfc = (const float*)d_in[4];   // (10,64,8,8)
    char* ws = (char*)d_ws;

    const size_t OFF_A   = 0;                       // 40 MB arena
    const size_t OFF_B   = 40ull << 20;             // 160 MB arena
    const size_t OFF_C   = OFF_B + (160ull << 20);  // u6 (96 KB)
    const size_t OFF_W2  = OFF_C + (1ull << 20);
    const size_t OFF_W3  = OFF_W2 + (1ull << 18);
    const size_t OFF_WFC = OFF_W3 + (1ull << 18);

    float*    p0  = (float*)(ws + OFF_A);
    float*    s1  = (float*)(ws + OFF_B);
    float*    p2  = (float*)(ws + OFF_A);   // reuses A (p0 dead)
    float*    p3  = (float*)(ws + OFF_B);   // reuses B (s1 dead)
    _Float16* ph3 = (_Float16*)(ws + OFF_A);
    float*    u3  = (float*)(ws + OFF_B);
    float*    p4  = (float*)(ws + OFF_A);
    float*    p5  = (float*)(ws + OFF_B);
    _Float16* ph5 = (_Float16*)(ws + OFF_A);
    float*    u5  = (float*)(ws + OFF_B);
    float*    p6  = (float*)(ws + OFF_A);
    _Float16* ph6 = (_Float16*)(ws + OFF_B);
    float*    u6  = (float*)(ws + OFF_C);
    _Float16* wp2  = (_Float16*)(ws + OFF_W2);
    _Float16* wp3  = (_Float16*)(ws + OFF_W3);
    _Float16* wpfc = (_Float16*)(ws + OFF_WFC);

    // 1. p0 = psp(x)                     rows: 8*2*34*34
    psp_fir<<<8 * 2 * 34 * 34, 128, 0, stream>>>(x, p0);
    // 2. s1 = spike(conv1(p0))           neurons: 8*16*32*32
    conv1_spike<<<(8 * 16 * 32 * 32) / 256, 256, 0, stream>>>(p0, w1, s1);
    // 3. p2 = psp(pool2(s1))             rows: 8*16*16*16
    pool_psp<<<8 * 16 * 16 * 16, 128, 0, stream>>>(s1, p2, 32, 32);
    // 4. s2 = spike(p2)  (in place)
    spike_scan<<<(8 * 16 * 16 * 16 + 255) / 256, 256, 0, stream>>>(p2, p2, 8 * 16 * 16 * 16);
    // 5. p3 = psp(s2)
    psp_fir<<<8 * 16 * 16 * 16, 128, 0, stream>>>(p2, p3);
    // 6-8. conv2 via WMMA: (8,16,16,16) -> (8,32,16,16)
    pack_act<<<8 * 16 * 16 * 16, 128, 0, stream>>>(p3, ph3, 16, 16, 16);
    pack_w<<<(32 * 160 + 255) / 256, 256, 0, stream>>>(w2, wp2, 32, 16, 3, 3, 32, 160);
    {
        int nT = 8 * 16 * 16 * 2 * 19;
        conv_wmma<16, 16, 16, 32, 32, 3, 3, 1, 16, 16, 160>
            <<<(nT * 32 + 255) / 256, 256, 0, stream>>>(ph3, wp2, u3, nT);
    }
    // 9. s3 = spike(u3) (in place)       rows: 8*32*16*16
    spike_scan<<<(8 * 32 * 16 * 16 + 255) / 256, 256, 0, stream>>>(u3, u3, 8 * 32 * 16 * 16);
    // 10. p4 = psp(pool2(s3))            rows: 8*32*8*8
    pool_psp<<<8 * 32 * 8 * 8, 128, 0, stream>>>(u3, p4, 16, 16);
    // 11. s4 = spike(p4) (in place)
    spike_scan<<<(8 * 32 * 8 * 8 + 255) / 256, 256, 0, stream>>>(p4, p4, 8 * 32 * 8 * 8);
    // 12. p5 = psp(s4)
    psp_fir<<<8 * 32 * 8 * 8, 128, 0, stream>>>(p4, p5);
    // 13-14. conv3 via WMMA: (8,32,8,8) -> (8,64,8,8)
    pack_act<<<8 * 32 * 8 * 8, 128, 0, stream>>>(p5, ph5, 32, 8, 8);
    pack_w<<<(64 * 288 + 255) / 256, 256, 0, stream>>>(w3, wp3, 64, 32, 3, 3, 64, 288);
    {
        int nT = 8 * 8 * 8 * 4 * 19;
        conv_wmma<32, 8, 8, 64, 64, 3, 3, 1, 8, 8, 288>
            <<<(nT * 32 + 255) / 256, 256, 0, stream>>>(ph5, wp3, u5, nT);
    }
    // 15. s5 = spike(u5) (in place)      rows: 8*64*8*8
    spike_scan<<<(8 * 64 * 8 * 8 + 255) / 256, 256, 0, stream>>>(u5, u5, 8 * 64 * 8 * 8);
    // 16. p6 = psp(s5)
    psp_fir<<<8 * 64 * 8 * 8, 128, 0, stream>>>(u5, p6);
    // 17-18. dense via WMMA: K = 64*8*8 = 4096, Co=10 padded to 16
    pack_act<<<8 * 64 * 8 * 8, 128, 0, stream>>>(p6, ph6, 64, 8, 8);
    pack_w<<<(16 * 4096 + 255) / 256, 256, 0, stream>>>(wfc, wpfc, 10, 64, 8, 8, 16, 4096);
    {
        int nT = 8 * 1 * 19;
        conv_wmma<64, 8, 8, 10, 16, 8, 8, 0, 1, 1, 4096>
            <<<(nT * 32 + 255) / 256, 256, 0, stream>>>(ph6, wpfc, u6, nT);
    }
    // 19. out = spike(u6)                rows: 8*10
    spike_scan<<<(80 + 255) / 256, 256, 0, stream>>>(u6, (float*)d_out, 80);
}
